// GraphBackboneGCN_40956808135086
// MI455X (gfx1250) — matrix operand order repaired and checked by reading, not compiled, not run on hardware
//
#include <hip/hip_runtime.h>

// ---------------------------------------------------------------------------
// GraphBackboneGCN for MI455X (gfx1250, wave32, WMMA)
//
// Pipeline per launch:
//   deg=1 -> atomic in-degree -> dinv=rsqrt(deg)
//   W1/2/3 -> bf16 transposed (for contiguous B-fragments)
//   3x [ h->bf16, WMMA GEMM (bf16 x bf16 -> f32), self-loop init,
//        coalesced edge scatter w/ native f32 atomics, bias+ReLU ]
//   atomic mean pool -> d_out
// ---------------------------------------------------------------------------

typedef __attribute__((ext_vector_type(16))) __bf16 v16bf;
typedef __attribute__((ext_vector_type(8)))  float  v8f;
typedef __attribute__((ext_vector_type(4)))  unsigned int v4u;
typedef __attribute__((ext_vector_type(4)))  float  v4f;

#define CDIV(a, b) (((a) + (b) - 1) / (b))

// ---------------------------- degree / dinv --------------------------------

__global__ __launch_bounds__(256) void k_deg_init(float* deg, int N) {
    int i = blockIdx.x * 256 + threadIdx.x;
    if (i < N) deg[i] = 1.0f;                 // self-loop contributes 1
}

__global__ __launch_bounds__(256) void k_deg_accum(const int* __restrict__ dst,
                                                   float* deg, int E) {
    int i = blockIdx.x * 256 + threadIdx.x;
    if (i < E) unsafeAtomicAdd(&deg[dst[i]], 1.0f);   // native global_atomic_add_f32
}

__global__ __launch_bounds__(256) void k_dinv(float* deg, int N) {
    int i = blockIdx.x * 256 + threadIdx.x;
    if (i < N) deg[i] = rsqrtf(deg[i]);       // in place: deg -> dinv
}

// ----------------------- weight convert (f32 -> bf16^T) --------------------

__global__ __launch_bounds__(256) void k_conv_w(const float* __restrict__ W,
                                                __bf16* __restrict__ Wt) {
    int i = blockIdx.x * 256 + threadIdx.x;   // i < 128*128
    if (i < 128 * 128) {
        int k = i >> 7, n = i & 127;
        Wt[n * 128 + k] = (__bf16)W[i];       // transpose: Wt[n][k] = W[k][n]
    }
}

__global__ __launch_bounds__(256) void k_conv_h(const float* __restrict__ h,
                                                __bf16* __restrict__ hb, int total) {
    int i = blockIdx.x * 256 + threadIdx.x;
    if (i < total) hb[i] = (__bf16)h[i];
}

// ------------------------------- WMMA GEMM ---------------------------------
// C[M x 128] = A[M x 128] (bf16) * W[128 x 128] (bf16, given transposed as Wt)
// One wave handles a 16-row M-tile across all 8 N-tiles; W staged in LDS.

union FragBF {
    v16bf v;
    v4u   u[2];
};

__global__ __launch_bounds__(256) void k_gemm_wmma(const __bf16* __restrict__ A,
                                                   const __bf16* __restrict__ Wt,
                                                   float* __restrict__ C, int M) {
    __shared__ __bf16 sW[128 * 128];          // 32 KB of the 320 KB WGP LDS
    {   // cooperative 128B-per-thread staging: global b128 -> ds b128
        const v4u* g = (const v4u*)Wt;
        v4u* l = (v4u*)sW;
        #pragma unroll
        for (int i = threadIdx.x; i < (128 * 128) / 8; i += 256) l[i] = g[i];
    }
    __syncthreads();

    const int wave = threadIdx.x >> 5;
    const int lane = threadIdx.x & 31;
    const int mtile = blockIdx.x * 8 + wave;
    if (mtile * 16 >= M) return;              // wave-uniform guard (EXEC all-1 in WMMA)
    const int m0 = mtile * 16;

    const int half = lane >> 4;               // 0: lanes 0-15, 1: lanes 16-31
    const int laneM = lane & 15;

    // A fragments: lane holds row m0+laneM, K chunks [kA,kA+8) and [kA+16,kA+24)
    const int kA = half * 8;
    FragBF a[4];
    #pragma unroll
    for (int kk = 0; kk < 4; ++kk) {
        const __bf16* p = A + (m0 + laneM) * 128 + kk * 32 + kA;
        a[kk].u[0] = *(const v4u*)(p);
        a[kk].u[1] = *(const v4u*)(p + 16);
    }

    const int kB = half * 8;                  // B: K chunks [kB,kB+8) and [kB+16,kB+24) per column
    #pragma unroll
    for (int nt = 0; nt < 8; ++nt) {
        v8f acc = {};
        const __bf16* bp = sW + (nt * 16 + laneM) * 128 + kB;
        #pragma unroll
        for (int kk = 0; kk < 4; ++kk) {
            FragBF b;
            b.u[0] = *(const v4u*)(bp + kk * 32);
            b.u[1] = *(const v4u*)(bp + kk * 32 + 16);
            acc = __builtin_amdgcn_wmma_f32_16x16x32_bf16(
                false, a[kk].v, false, b.v, (short)0, acc, false, false);
        }
        // C layout: VGPR r -> M = m0 + r + 8*half, N = nt*16 + laneM
        float* cp = C + (m0 + half * 8) * 128 + nt * 16 + laneM;
        #pragma unroll
        for (int r = 0; r < 8; ++r) cp[r * 128] = acc[r];
    }
}

// --------------------------- aggregation stage -----------------------------

// agg[i,:] = t[i,:] * dinv[i]^2   (self-loop term; also full overwrite/init)
__global__ __launch_bounds__(256) void k_agg_init(const float* __restrict__ t,
                                                  const float* __restrict__ dinv,
                                                  float* __restrict__ agg, int total) {
    int i = blockIdx.x * 256 + threadIdx.x;
    if (i < total) {
        float di = dinv[i >> 7];
        agg[i] = t[i] * di * di;
    }
}

// One wave per edge: 32 lanes x float4 = one coalesced 512B row gather,
// then 4 native f32 atomics per lane into the destination row.
__global__ __launch_bounds__(256) void k_edge_scatter(const float* __restrict__ t,
                                                      const float* __restrict__ dinv,
                                                      const int* __restrict__ src,
                                                      const int* __restrict__ dst,
                                                      float* __restrict__ agg, int E) {
    int e = blockIdx.x * 8 + (threadIdx.x >> 5);
    if (e >= E) return;
    int lane = threadIdx.x & 31;
    int s = src[e], d = dst[e];
    float c = dinv[s] * dinv[d];
    v4f v = *(const v4f*)(t + s * 128 + lane * 4);
    float* o = agg + d * 128 + lane * 4;
    unsafeAtomicAdd(o + 0, v.x * c);
    unsafeAtomicAdd(o + 1, v.y * c);
    unsafeAtomicAdd(o + 2, v.z * c);
    unsafeAtomicAdd(o + 3, v.w * c);
}

__global__ __launch_bounds__(256) void k_bias_relu(float* __restrict__ h,
                                                   const float* __restrict__ b, int total) {
    int i = blockIdx.x * 256 + threadIdx.x;
    if (i < total) {
        float v = h[i] + b[i & 127];
        h[i] = v > 0.0f ? v : 0.0f;
    }
}

// ------------------------------- pooling -----------------------------------

__global__ __launch_bounds__(256) void k_pool_zero(float* __restrict__ out,
                                                   float* __restrict__ cnt) {
    int i = blockIdx.x * 256 + threadIdx.x;
    if (i < 512 * 128) out[i] = 0.0f;
    if (i < 512) cnt[i] = 0.0f;
}

__global__ __launch_bounds__(256) void k_node_pool(const float* __restrict__ h,
                                                   const int* __restrict__ batch,
                                                   float* __restrict__ out,
                                                   float* __restrict__ cnt, int N) {
    int n = blockIdx.x * 8 + (threadIdx.x >> 5);
    if (n >= N) return;
    int lane = threadIdx.x & 31;
    int g = batch[n];
    v4f v = *(const v4f*)(h + n * 128 + lane * 4);
    float* o = out + g * 128 + lane * 4;
    unsafeAtomicAdd(o + 0, v.x);
    unsafeAtomicAdd(o + 1, v.y);
    unsafeAtomicAdd(o + 2, v.z);
    unsafeAtomicAdd(o + 3, v.w);
    if (lane == 0) unsafeAtomicAdd(&cnt[g], 1.0f);
}

__global__ __launch_bounds__(256) void k_pool_div(float* __restrict__ out,
                                                  const float* __restrict__ cnt) {
    int i = blockIdx.x * 256 + threadIdx.x;
    if (i < 512 * 128) {
        float c = cnt[i >> 7];
        out[i] = out[i] / (c > 1.0f ? c : 1.0f);
    }
}

// ------------------------------- launcher ----------------------------------

extern "C" void kernel_launch(void* const* d_in, const int* in_sizes, int n_in,
                              void* d_out, int out_size, void* d_ws, size_t ws_size,
                              hipStream_t stream) {
    const float* x  = (const float*)d_in[0];
    // d_in[1] = edge_attr (unused by GCNConv, matching reference)
    const float* W[3] = {(const float*)d_in[2], (const float*)d_in[4], (const float*)d_in[6]};
    const float* B[3] = {(const float*)d_in[3], (const float*)d_in[5], (const float*)d_in[7]};
    const int*   ei    = (const int*)d_in[8];
    const int*   batch = (const int*)d_in[9];

    const int N = in_sizes[0] / 128;
    const int E = in_sizes[8] / 2;
    const int NF = N * 128;
    const int* src = ei;
    const int* dst = ei + E;

    // workspace carve-up (256B aligned)
    char* base = (char*)d_ws;
    size_t off = 0;
    auto carve = [&](size_t bytes) -> char* {
        char* p = base + off;
        off = (off + bytes + 255) & ~(size_t)255;
        return p;
    };
    float*  dinv  = (float*)carve((size_t)N * 4);          // deg, then dinv in place
    __bf16* WtB   = (__bf16*)carve((size_t)3 * 128 * 128 * 2);
    __bf16* hb    = (__bf16*)carve((size_t)NF * 2);
    float*  t     = (float*)carve((size_t)NF * 4);
    float*  aggA  = (float*)carve((size_t)NF * 4);
    float*  aggB  = (float*)carve((size_t)NF * 4);
    float*  cnt   = (float*)carve((size_t)512 * 4);
    (void)ws_size;

    float* outp = (float*)d_out;

    // --- degree normalization ---
    k_deg_init <<<CDIV(N, 256), 256, 0, stream>>>(dinv, N);
    k_deg_accum<<<CDIV(E, 256), 256, 0, stream>>>(dst, dinv, E);
    k_dinv     <<<CDIV(N, 256), 256, 0, stream>>>(dinv, N);

    // --- weights -> bf16 transposed ---
    for (int l = 0; l < 3; ++l)
        k_conv_w<<<CDIV(128 * 128, 256), 256, 0, stream>>>(W[l], WtB + (size_t)l * 128 * 128);

    // --- three GCN layers ---
    const int gemmBlocks = CDIV(CDIV(N, 16), 8);
    const float* hin = x;
    float* hout[3] = {aggA, aggB, aggA};
    for (int l = 0; l < 3; ++l) {
        float* agg = hout[l];
        k_conv_h      <<<CDIV(NF, 256), 256, 0, stream>>>(hin, hb, NF);
        k_gemm_wmma   <<<gemmBlocks, 256, 0, stream>>>(hb, WtB + (size_t)l * 128 * 128, t, N);
        k_agg_init    <<<CDIV(NF, 256), 256, 0, stream>>>(t, dinv, agg, NF);
        k_edge_scatter<<<CDIV(E, 8), 256, 0, stream>>>(t, dinv, src, dst, agg, E);
        k_bias_relu   <<<CDIV(NF, 256), 256, 0, stream>>>(agg, B[l], NF);
        hin = agg;
    }

    // --- global mean pool ---
    k_pool_zero<<<CDIV(512 * 128, 256), 256, 0, stream>>>(outp, cnt);
    k_node_pool<<<CDIV(N, 8), 256, 0, stream>>>(hout[2], batch, outp, cnt, N);
    k_pool_div <<<CDIV(512 * 128, 256), 256, 0, stream>>>(outp, cnt);
}
